// TemporalGAT_1374389534858
// MI455X (gfx1250) — compile-verified, matrix operand order checked by reading, not run on hardware
//
#include <hip/hip_runtime.h>
#include <hip/hip_bf16.h>

// ---------------- problem constants (match reference) ----------------
#define NN     50000
#define EE     800000
#define ETOT   (EE + NN)        // edges + self loops = 850000
#define F_INN  64
#define HIDD   32
#define HEADS  4
#define HD     (HEADS * HIDD)   // 128
#define BB     16
#define OUTC   8
#define SLOPE  0.2f

typedef _Float16 half16 __attribute__((ext_vector_type(16)));
typedef _Float16 half8  __attribute__((ext_vector_type(8)));
typedef float    float8 __attribute__((ext_vector_type(8)));

// ---------------- small utility kernels ----------------
__global__ void k_fill(float* p, float v, int n) {
  int i = blockIdx.x * blockDim.x + threadIdx.x;
  if (i < n) p[i] = v;
}

__global__ void k_cast16(const float* __restrict__ s, _Float16* __restrict__ d, int n) {
  int i = blockIdx.x * blockDim.x + threadIdx.x;
  if (i < n) d[i] = (_Float16)s[i];
}

// Wt[n*K + k] = (f16) W[k*Nc + n]  (transpose so B columns are contiguous)
__global__ void k_castT(const float* __restrict__ W, _Float16* __restrict__ Wt, int K, int Nc) {
  int i = blockIdx.x * blockDim.x + threadIdx.x;
  if (i < K * Nc) {
    int k = i / Nc, n = i % Nc;
    Wt[n * K + k] = (_Float16)W[i];
  }
}

__global__ void k_addbias(const float* __restrict__ a, const float* __restrict__ b,
                          float* __restrict__ c, int n) {
  int i = blockIdx.x * blockDim.x + threadIdx.x;
  if (i < n) c[i] = a[i] + b[i];
}

// ---------------- WMMA GEMM: C = act(A[M,K] @ B[K,Nc] + bias) ----------------
// A: f16 row-major [M,K];  Bt: f16 [Nc,K] (B transposed);  C: f32 [M,Nc]
// One wave computes one 16x16 output tile via K/32 v_wmma_f32_16x16x32_f16 ops.
// 16-bit A layout (ISA 7.12.2): lane = {half = L>>4, r = L&15}; row/col index = r;
//   VGPR 0..3 hold K = {2j, 2j+1} + half*8;  VGPR 4..7 hold K = 16 + {...} + half*8
// => per 32-K chunk each lane loads two contiguous half8 (16B) slabs.
__global__ void k_gemm_wmma(const _Float16* __restrict__ A, const _Float16* __restrict__ Bt,
                            const float* __restrict__ bias,
                            float* __restrict__ C, _Float16* __restrict__ C16,
                            int M, int K, int Nc, int act) {
  int gw   = (blockIdx.x * blockDim.x + threadIdx.x) >> 5;  // global wave id
  int lane = threadIdx.x & 31;
  int ntiles = Nc >> 4, mtiles = M >> 4;
  int mt = gw / ntiles, nt = gw % ntiles;
  if (mt >= mtiles) return;
  int hi = lane >> 4;   // which 16-lane half
  int r  = lane & 15;   // row (A) / column (B) index within tile

  const _Float16* Arow = A  + (size_t)(mt * 16 + r) * K;
  const _Float16* Brow = Bt + (size_t)(nt * 16 + r) * K;

  float8 acc = (float8)0.0f;
  for (int k0 = 0; k0 < K; k0 += 32) {
    half8 alo = *(const half8*)(Arow + k0 + hi * 8);
    half8 ahi = *(const half8*)(Arow + k0 + 16 + hi * 8);
    half8 blo = *(const half8*)(Brow + k0 + hi * 8);
    half8 bhi = *(const half8*)(Brow + k0 + 16 + hi * 8);
    half16 a, b;
#pragma unroll
    for (int i = 0; i < 8; ++i) {
      a[i] = alo[i]; a[i + 8] = ahi[i];
      b[i] = blo[i]; b[i + 8] = bhi[i];
    }
    acc = __builtin_amdgcn_wmma_f32_16x16x32_f16(false, a, false, b,
                                                 (short)0, acc, false, false);
  }

  // C/D layout: VGPR i -> M = i + hi*8, N = r
#pragma unroll
  for (int i = 0; i < 8; ++i) {
    int m = mt * 16 + i + hi * 8;
    int n = nt * 16 + r;
    float v = acc[i] + bias[n];
    if (act == 1) v = fmaxf(v, 0.0f);
    size_t o = (size_t)m * Nc + n;
    C[o] = v;
    if (C16) C16[o] = (_Float16)v;
  }
}

// ---------------- edge / attention kernels ----------------
__device__ inline void atomicMaxF(float* addr, float val) {
  if (val >= 0.0f) atomicMax((int*)addr, __float_as_int(val));
  else             atomicMin((unsigned int*)addr, (unsigned int)__float_as_int(val));
}

__device__ inline void edge_ends(const int* __restrict__ ei, int e, int& s, int& d) {
  if (e < EE) { s = ei[e]; d = ei[EE + e]; }
  else        { s = e - EE; d = e - EE; }   // self loops appended
}

// pass 1: score[e,h] = <leakyrelu(xl[src]+xr[dst]), att[h]>; segment max over dst
__global__ void k_score(const int* __restrict__ ei, const float* __restrict__ xl,
                        const float* __restrict__ xr, const float* __restrict__ att,
                        float* __restrict__ score, float* __restrict__ smax) {
  int idx = blockIdx.x * blockDim.x + threadIdx.x;
  if (idx >= ETOT * HEADS) return;
  int e = idx >> 2, h = idx & 3, s, d;
  edge_ends(ei, e, s, d);
  const float4* pl = (const float4*)(xl + (size_t)s * HD + h * HIDD);
  const float4* pr = (const float4*)(xr + (size_t)d * HD + h * HIDD);
  const float4* pa = (const float4*)(att + h * HIDD);
  float sc = 0.0f;
#pragma unroll
  for (int i = 0; i < HIDD / 4; ++i) {
    float4 a = pl[i], b = pr[i], w = pa[i];
    float vx = a.x + b.x; vx = vx > 0.0f ? vx : SLOPE * vx;
    float vy = a.y + b.y; vy = vy > 0.0f ? vy : SLOPE * vy;
    float vz = a.z + b.z; vz = vz > 0.0f ? vz : SLOPE * vz;
    float vw = a.w + b.w; vw = vw > 0.0f ? vw : SLOPE * vw;
    sc += vx * w.x + vy * w.y + vz * w.z + vw * w.w;
  }
  score[idx] = sc;
  atomicMaxF(smax + (size_t)d * HEADS + h, sc);
}

// pass 2: ex = exp(score - smax[dst]); segment sum over dst
__global__ void k_expsum(const int* __restrict__ ei, const float* __restrict__ score,
                         const float* __restrict__ smax, float* __restrict__ ex,
                         float* __restrict__ den) {
  int idx = blockIdx.x * blockDim.x + threadIdx.x;
  if (idx >= ETOT * HEADS) return;
  int e = idx >> 2, h = idx & 3, s, d;
  edge_ends(ei, e, s, d);
  float v = __expf(score[idx] - smax[(size_t)d * HEADS + h]);
  ex[idx] = v;
  atomicAdd(den + (size_t)d * HEADS + h, v);
}

// pass 3: alpha = ex/den; agg[dst] += alpha * xl[src]
__global__ void k_agg(const int* __restrict__ ei, const float* __restrict__ ex,
                      const float* __restrict__ den, const float* __restrict__ xl,
                      float* __restrict__ alphaOut, float* __restrict__ agg) {
  int idx = blockIdx.x * blockDim.x + threadIdx.x;
  if (idx >= ETOT * HEADS) return;
  int e = idx >> 2, h = idx & 3, s, d;
  edge_ends(ei, e, s, d);
  float a = ex[idx] / (den[(size_t)d * HEADS + h] + 1e-16f);
  alphaOut[idx] = a;
  const float4* pl = (const float4*)(xl + (size_t)s * HD + h * HIDD);
  float* pd = agg + (size_t)d * HD + h * HIDD;
#pragma unroll
  for (int i = 0; i < HIDD / 4; ++i) {
    float4 v = pl[i];
    atomicAdd(pd + 4 * i + 0, a * v.x);
    atomicAdd(pd + 4 * i + 1, a * v.y);
    atomicAdd(pd + 4 * i + 2, a * v.z);
    atomicAdd(pd + 4 * i + 3, a * v.w);
  }
}

// layers 0/1: h = elu(agg + bias), concat heads (already concatenated in agg)
__global__ void k_finish_concat(const float* __restrict__ agg, const float* __restrict__ bias,
                                float* __restrict__ h32, _Float16* __restrict__ h16) {
  int idx = blockIdx.x * blockDim.x + threadIdx.x;
  if (idx >= NN * HD) return;
  int j = idx & (HD - 1);
  float v = agg[idx] + bias[j];
  v = v > 0.0f ? v : (__expf(v) - 1.0f);   // elu
  h32[idx] = v;
  h16[idx] = (_Float16)v;
}

// layer 2: h = mean over heads + bias;  node_emb output, f16 copy for LSTM GEMM
__global__ void k_finish_mean(const float* __restrict__ agg, const float* __restrict__ bias2,
                              float* __restrict__ node_emb, _Float16* __restrict__ h16) {
  int idx = blockIdx.x * blockDim.x + threadIdx.x;
  if (idx >= NN * HIDD) return;
  int n = idx >> 5, c = idx & 31;
  const float* p = agg + (size_t)n * HD;
  float v = 0.25f * (p[c] + p[HIDD + c] + p[2 * HIDD + c] + p[3 * HIDD + c]) + bias2[c];
  node_emb[idx] = v;
  h16[idx] = (_Float16)v;
}

// single-step LSTM with zero initial state: c = sig(i)*tanh(g), ht = sig(o)*tanh(c)
// plus batch-segment mean-pool accumulation
__global__ void k_lstm(const float* __restrict__ gates, const int* __restrict__ batch,
                       float* __restrict__ pool, float* __restrict__ cnt) {
  int idx = blockIdx.x * blockDim.x + threadIdx.x;
  if (idx >= NN * HIDD) return;
  int n = idx >> 5, c = idx & 31;
  const float* g = gates + (size_t)n * (4 * HIDD);
  float ig = g[c];
  float gg = g[2 * HIDD + c];
  float og = g[3 * HIDD + c];
  float cs = (1.0f / (1.0f + __expf(-ig))) * tanhf(gg);
  float ht = (1.0f / (1.0f + __expf(-og))) * tanhf(cs);
  int b = batch[n];
  atomicAdd(pool + (size_t)b * HIDD + c, ht);
  if (c == 0) atomicAdd(cnt + b, 1.0f);
}

// out[b,o] = (pool[b]/cnt[b]) @ W_out + b_out   (16x32 @ 32x8, one block)
__global__ void k_out(const float* __restrict__ pool, const float* __restrict__ cnt,
                      const float* __restrict__ W_out, const float* __restrict__ b_out,
                      float* __restrict__ out) {
  int t = threadIdx.x;
  if (t >= BB * OUTC) return;
  int b = t / OUTC, o = t % OUTC;
  float inv = 1.0f / fmaxf(cnt[b], 1.0f);
  float s = b_out[o];
  for (int c = 0; c < HIDD; ++c)
    s += (pool[b * HIDD + c] * inv) * W_out[c * OUTC + o];
  out[t] = s;
}

// ---------------- host orchestration ----------------
static inline int cdiv(long a, long b) { return (int)((a + b - 1) / b); }

extern "C" void kernel_launch(void* const* d_in, const int* in_sizes, int n_in,
                              void* d_out, int out_size, void* d_ws, size_t ws_size,
                              hipStream_t stream) {
  (void)in_sizes; (void)n_in; (void)out_size; (void)ws_size;

  const float* x     = (const float*)d_in[0];
  const int*   ei    = (const int*)d_in[1];
  const int*   batch = (const int*)d_in[2];
  const float* W_in  = (const float*)d_in[3];
  const float* b_in  = (const float*)d_in[4];
  const float* Wl[3]   = {(const float*)d_in[5],  (const float*)d_in[11], (const float*)d_in[17]};
  const float* Wr[3]   = {(const float*)d_in[6],  (const float*)d_in[12], (const float*)d_in[18]};
  const float* bl[3]   = {(const float*)d_in[7],  (const float*)d_in[13], (const float*)d_in[19]};
  const float* br[3]   = {(const float*)d_in[8],  (const float*)d_in[14], (const float*)d_in[20]};
  const float* att[3]  = {(const float*)d_in[9],  (const float*)d_in[15], (const float*)d_in[21]};
  const float* bias[3] = {(const float*)d_in[10], (const float*)d_in[16], (const float*)d_in[22]};
  const float* W_ih  = (const float*)d_in[23];
  const float* b_ih  = (const float*)d_in[25];
  const float* b_hh  = (const float*)d_in[26];
  const float* W_out = (const float*)d_in[27];
  const float* b_out = (const float*)d_in[28];

  // workspace bump allocator (256B aligned)
  char* wp = (char*)d_ws;
  auto alloc = [&](size_t bytes) -> char* {
    char* r = wp; wp += (bytes + 255) & ~(size_t)255; return r;
  };
  int din[3] = {HIDD, HD, HD};
  _Float16* X16  = (_Float16*)alloc((size_t)NN * F_INN * 2);
  _Float16* WinT = (_Float16*)alloc((size_t)F_INN * HIDD * 2);
  _Float16 *WlT[3], *WrT[3];
  for (int i = 0; i < 3; ++i) {
    WlT[i] = (_Float16*)alloc((size_t)din[i] * HD * 2);
    WrT[i] = (_Float16*)alloc((size_t)din[i] * HD * 2);
  }
  _Float16* WihT  = (_Float16*)alloc((size_t)HIDD * 4 * HIDD * 2);
  float*    bcomb = (float*)alloc(4 * HIDD * 4);
  float*    H32   = (float*)alloc((size_t)NN * HD * 4);
  _Float16* H16   = (_Float16*)alloc((size_t)NN * HD * 2);
  float*    XL    = (float*)alloc((size_t)NN * HD * 4);
  float*    XR    = (float*)alloc((size_t)NN * HD * 4);
  float*    SCOR  = (float*)alloc((size_t)ETOT * HEADS * 4);
  float*    EXV   = (float*)alloc((size_t)ETOT * HEADS * 4);
  float*    SMAX  = (float*)alloc((size_t)NN * HEADS * 4);
  float*    DEN   = (float*)alloc((size_t)NN * HEADS * 4);
  float*    AGG   = (float*)alloc((size_t)NN * HD * 4);
  float*    GATES = XL;   // reuse: XL dead once last aggregation done
  float*    POOL  = (float*)alloc((size_t)BB * HIDD * 4);
  float*    CNT   = (float*)alloc((size_t)BB * 4);

  // output regions: (out, node_emb, alpha0, alpha1, alpha2) flat concat
  float* out      = (float*)d_out;
  float* node_emb = out + BB * OUTC;
  float* alpha0   = node_emb + (size_t)NN * HIDD;
  float* alpha1   = alpha0 + (size_t)ETOT * HEADS;
  float* alpha2   = alpha1 + (size_t)ETOT * HEADS;
  float* alphas[3] = {alpha0, alpha1, alpha2};

  const int BLK = 256;

  // --- precision conversions (weights tiny, x once) ---
  k_cast16<<<cdiv((long)NN * F_INN, BLK), BLK, 0, stream>>>(x, X16, NN * F_INN);
  k_castT <<<cdiv(F_INN * HIDD, BLK), BLK, 0, stream>>>(W_in, WinT, F_INN, HIDD);
  for (int i = 0; i < 3; ++i) {
    k_castT<<<cdiv(din[i] * HD, BLK), BLK, 0, stream>>>(Wl[i], WlT[i], din[i], HD);
    k_castT<<<cdiv(din[i] * HD, BLK), BLK, 0, stream>>>(Wr[i], WrT[i], din[i], HD);
  }
  k_castT<<<cdiv(HIDD * 4 * HIDD, BLK), BLK, 0, stream>>>(W_ih, WihT, HIDD, 4 * HIDD);
  k_addbias<<<1, 4 * HIDD, 0, stream>>>(b_ih, b_hh, bcomb, 4 * HIDD);

  // --- input projection: h = relu(x @ W_in + b_in) ---
  {
    long tiles = (NN / 16) * (HIDD / 16);
    k_gemm_wmma<<<cdiv(tiles * 32, BLK), BLK, 0, stream>>>(
        X16, WinT, b_in, H32, H16, NN, F_INN, HIDD, 1);
  }

  // --- 3 GATv2 layers ---
  for (int L = 0; L < 3; ++L) {
    k_fill<<<cdiv(NN * HEADS, BLK), BLK, 0, stream>>>(SMAX, -__builtin_huge_valf(), NN * HEADS);
    k_fill<<<cdiv(NN * HEADS, BLK), BLK, 0, stream>>>(DEN, 0.0f, NN * HEADS);
    k_fill<<<cdiv(NN * HD, BLK), BLK, 0, stream>>>(AGG, 0.0f, NN * HD);

    long tiles = (NN / 16) * (HD / 16);
    k_gemm_wmma<<<cdiv(tiles * 32, BLK), BLK, 0, stream>>>(
        H16, WlT[L], bl[L], XL, (_Float16*)nullptr, NN, din[L], HD, 0);
    k_gemm_wmma<<<cdiv(tiles * 32, BLK), BLK, 0, stream>>>(
        H16, WrT[L], br[L], XR, (_Float16*)nullptr, NN, din[L], HD, 0);

    long eth = (long)ETOT * HEADS;
    k_score <<<cdiv(eth, BLK), BLK, 0, stream>>>(ei, XL, XR, att[L], SCOR, SMAX);
    k_expsum<<<cdiv(eth, BLK), BLK, 0, stream>>>(ei, SCOR, SMAX, EXV, DEN);
    k_agg   <<<cdiv(eth, BLK), BLK, 0, stream>>>(ei, EXV, DEN, XL, alphas[L], AGG);

    if (L < 2)
      k_finish_concat<<<cdiv(NN * HD, BLK), BLK, 0, stream>>>(AGG, bias[L], H32, H16);
    else
      k_finish_mean<<<cdiv(NN * HIDD, BLK), BLK, 0, stream>>>(AGG, bias[L], node_emb, H16);
  }

  // --- LSTM gates GEMM + cell + batch pooling ---
  {
    long tiles = (NN / 16) * ((4 * HIDD) / 16);
    k_gemm_wmma<<<cdiv(tiles * 32, BLK), BLK, 0, stream>>>(
        H16, WihT, bcomb, GATES, (_Float16*)nullptr, NN, HIDD, 4 * HIDD, 0);
  }
  k_fill<<<1, BLK, 0, stream>>>(POOL, 0.0f, BB * HIDD);
  k_fill<<<1, BB, 0, stream>>>(CNT, 0.0f, BB);
  k_lstm<<<cdiv(NN * HIDD, BLK), BLK, 0, stream>>>(GATES, batch, POOL, CNT);
  k_out<<<1, 128, 0, stream>>>(POOL, CNT, W_out, b_out, out);
}
